// DeformSpaceAttentionv5_26079041421812
// MI455X (gfx1250) — compile-verified, matrix operand order checked by reading, not run on hardware
//
#include <hip/hip_runtime.h>
#include <hip/hip_bf16.h>

#define B_   4
#define C_   256
#define H_   128
#define W_   128
#define HW_  (H_ * W_)
#define NPIX (B_ * HW_)

typedef __attribute__((ext_vector_type(16))) __bf16 v16bf;
typedef __attribute__((ext_vector_type(8)))  float  v8f;

// ---------------------------------------------------------------------------
// Fragment loader: row-major bf16 matrix with row stride 256.
// CDNA5 16-bit 16x32 layout (ISA 7.12.2): lane half = lane>>4,
//   elements 0..7  -> K = kt*32 + half*8 + e        (16 contiguous bytes)
//   elements 8..15 -> K = kt*32 + 16 + half*8 + e'  (16 contiguous bytes, +32 B)
// ---------------------------------------------------------------------------
__device__ inline v16bf load_frag(const __hip_bfloat16* __restrict__ base,
                                  int row, int kt, int half) {
    union { v16bf v; uint4 u[2]; } f;
    const char* p = (const char*)(base + (size_t)row * 256 + kt * 32 + half * 8);
    f.u[0] = *(const uint4*)(p);
    f.u[1] = *(const uint4*)(p + 32);   // +16 bf16 elements
    return f.v;
}

// ---------------------------------------------------------------------------
// Kernel 1: convert both weight matrices f32 -> bf16
// ---------------------------------------------------------------------------
__global__ __launch_bounds__(256)
void conv_w_kernel(const float* __restrict__ w0, const float* __restrict__ w1,
                   __hip_bfloat16* __restrict__ w0b, __hip_bfloat16* __restrict__ w1b) {
    int i = blockIdx.x * 256 + threadIdx.x;
    if (i < C_ * C_) {
        w0b[i] = __float2bfloat16(w0[i]);
        w1b[i] = __float2bfloat16(w1[i]);
    }
}

// ---------------------------------------------------------------------------
// Kernel 2: transpose-convert x [b][c][p] f32 -> x_t [b*HW+p][c] bf16
// 64ch x 64px LDS tile, pad 65 to kill bank conflicts.
// ---------------------------------------------------------------------------
__global__ __launch_bounds__(256)
void transpose_x_kernel(const float* __restrict__ x, __hip_bfloat16* __restrict__ xt) {
    __shared__ float tile[64 * 65];
    int bid = blockIdx.x;               // B*(C/64)*(HW/64) = 4*4*256 = 4096
    int b   = bid >> 10;
    int rem = bid & 1023;
    int c0  = (rem >> 8) << 6;
    int p0  = (rem & 255) << 6;
    const float* src = x + ((size_t)(b * C_ + c0)) * HW_ + p0;
#pragma unroll
    for (int i = 0; i < 16; ++i) {
        int idx = i * 256 + threadIdx.x;
        int c = idx >> 6, p = idx & 63;
        tile[c * 65 + p] = src[(size_t)c * HW_ + p];
    }
    __syncthreads();
    __hip_bfloat16* dst = xt + ((size_t)(b * HW_ + p0)) * 256 + c0;
#pragma unroll
    for (int i = 0; i < 16; ++i) {
        int idx = i * 256 + threadIdx.x;
        int p = idx >> 6, c = idx & 63;
        dst[(size_t)p * 256 + c] = __float2bfloat16(tile[c * 65 + p]);
    }
}

// ---------------------------------------------------------------------------
// Kernel 3: deformable 3x3 bilinear max-unfold -> q_t [b*HW+p][c] bf16
// One thread = one pixel x 32 channels. Corner indices/weights precomputed.
// ---------------------------------------------------------------------------
__global__ __launch_bounds__(256)
void deform_max_kernel(const float* __restrict__ x, const float* __restrict__ off,
                       __hip_bfloat16* __restrict__ qt) {
    int gid = blockIdx.x * 256 + threadIdx.x;      // NPIX * 8 threads
    int grp = gid / NPIX;                          // channel group 0..7
    int pid = gid - grp * NPIX;                    // global pixel
    int b = pid / HW_;
    int p = pid - b * HW_;
    int h = p / W_;
    int w = p - h * W_;

    const float* offb = off + (size_t)b * 18 * HW_ + p;

    int   idxs[9][4];
    float wts[9][4];
#pragma unroll
    for (int k = 0; k < 9; ++k) {
        float oy = offb[(size_t)(2 * k)     * HW_];
        float ox = offb[(size_t)(2 * k + 1) * HW_];
        float py = (float)(h + k / 3 - 1) + oy;
        float px = (float)(w + k % 3 - 1) + ox;
        float y0f = floorf(py), x0f = floorf(px);
        float wy = py - y0f, wx = px - x0f;
        int y0 = (int)y0f, x0 = (int)x0f;
#pragma unroll
        for (int j = 0; j < 4; ++j) {
            int yy = y0 + (j >> 1);
            int xx = x0 + (j & 1);
            bool valid = (yy >= 0) && (yy < H_) && (xx >= 0) && (xx < W_);
            int yc = min(max(yy, 0), H_ - 1);
            int xc = min(max(xx, 0), W_ - 1);
            idxs[k][j] = yc * W_ + xc;
            float wj = ((j >> 1) ? wy : 1.f - wy) * ((j & 1) ? wx : 1.f - wx);
            wts[k][j] = valid ? wj : 0.f;
        }
    }

    const float* xb = x + (size_t)b * C_ * HW_;
    __hip_bfloat16* qrow = qt + (size_t)pid * 256;
    for (int ci = 0; ci < 32; ++ci) {
        int c = grp * 32 + ci;
        const float* xc = xb + (size_t)c * HW_;
        float m = -3.4e38f;
#pragma unroll
        for (int k = 0; k < 9; ++k) {
            float s = wts[k][0] * xc[idxs[k][0]] + wts[k][1] * xc[idxs[k][1]]
                    + wts[k][2] * xc[idxs[k][2]] + wts[k][3] * xc[idxs[k][3]];
            m = fmaxf(m, s);
        }
        qrow[c] = __float2bfloat16(m);
    }
}

// ---------------------------------------------------------------------------
// Kernel 4: fused dual-GEMM (query = w0*q, key = w1*x) + channel-norm dot.
// 1 block (8 waves) per 16-pixel tile. Wave wv handles M-tiles {wv, wv+8}
// of both GEMMs over K=256 (8 steps of v_wmma_f32_16x16x32_bf16).
// Epilogue: per-pixel moments S1,S2,K1,K2,S12 via ds_add_f32, then
//   out = (S12 - S1*K1/C) / sqrt((S2 - S1^2/C + eps)*(K2 - K1^2/C + eps))
// ---------------------------------------------------------------------------
__global__ __launch_bounds__(256)
void fused_gemm_kernel(const __hip_bfloat16* __restrict__ qt,
                       const __hip_bfloat16* __restrict__ xt,
                       const __hip_bfloat16* __restrict__ w0b,
                       const __hip_bfloat16* __restrict__ w1b,
                       const float* __restrict__ b0, const float* __restrict__ b1,
                       float* __restrict__ out) {
    __shared__ float red[5][16];
    int tid = threadIdx.x;
    if (tid < 80) (&red[0][0])[tid] = 0.f;
    __syncthreads();

    int g    = blockIdx.x * 16;       // global pixel row base (flat over B*HW)
    int lane = tid & 31;
    int wv   = tid >> 5;              // wave 0..7
    int half = lane >> 4;
    int nm   = lane & 15;

    int m0 = wv;                      // M-tile rows m*16 .. m*16+15
    int m1 = wv + 8;

    v8f cq0 = {0.f,0.f,0.f,0.f,0.f,0.f,0.f,0.f};
    v8f cq1 = cq0, ck0 = cq0, ck1 = cq0;

#pragma unroll
    for (int kt = 0; kt < 8; ++kt) {
        v16bf bq = load_frag(qt, g + nm, kt, half);
        v16bf bx = load_frag(xt, g + nm, kt, half);
        v16bf a;
        a = load_frag(w0b, m0 * 16 + nm, kt, half);
        cq0 = __builtin_amdgcn_wmma_f32_16x16x32_bf16(false, a, false, bq, (short)0, cq0, false, false);
        a = load_frag(w1b, m0 * 16 + nm, kt, half);
        ck0 = __builtin_amdgcn_wmma_f32_16x16x32_bf16(false, a, false, bx, (short)0, ck0, false, false);
        a = load_frag(w0b, m1 * 16 + nm, kt, half);
        cq1 = __builtin_amdgcn_wmma_f32_16x16x32_bf16(false, a, false, bq, (short)0, cq1, false, false);
        a = load_frag(w1b, m1 * 16 + nm, kt, half);
        ck1 = __builtin_amdgcn_wmma_f32_16x16x32_bf16(false, a, false, bx, (short)0, ck1, false, false);
    }

    // Per-lane partial moments for pixel nm (lanes l and l+16 cover
    // complementary o-halves of the same pixel column).
    float s1 = 0.f, s2 = 0.f, k1 = 0.f, k2 = 0.f, s12 = 0.f;
    int ob0 = m0 * 16 + half * 8;
    int ob1 = m1 * 16 + half * 8;
#pragma unroll
    for (int r = 0; r < 8; ++r) {
        float q = cq0[r] + b0[ob0 + r];
        float k = ck0[r] + b1[ob0 + r];
        s1 += q; s2 += q * q; k1 += k; k2 += k * k; s12 += q * k;
        q = cq1[r] + b0[ob1 + r];
        k = ck1[r] + b1[ob1 + r];
        s1 += q; s2 += q * q; k1 += k; k2 += k * k; s12 += q * k;
    }
    atomicAdd(&red[0][nm], s1);
    atomicAdd(&red[1][nm], s2);
    atomicAdd(&red[2][nm], k1);
    atomicAdd(&red[3][nm], k2);
    atomicAdd(&red[4][nm], s12);
    __syncthreads();

    if (tid < 16) {
        float S1  = red[0][tid], S2 = red[1][tid];
        float K1  = red[2][tid], K2 = red[3][tid];
        float S12 = red[4][tid];
        const float inv = 1.0f / 256.0f;
        float vq  = S2 - S1 * S1 * inv + 1e-5f;
        float vk  = K2 - K1 * K1 * inv + 1e-5f;
        float num = S12 - S1 * K1 * inv;
        out[g + tid] = num * __frsqrt_rn(vq * vk);
    }
}

// ---------------------------------------------------------------------------
extern "C" void kernel_launch(void* const* d_in, const int* in_sizes, int n_in,
                              void* d_out, int out_size, void* d_ws, size_t ws_size,
                              hipStream_t stream) {
    (void)in_sizes; (void)n_in; (void)out_size; (void)ws_size;
    const float* x      = (const float*)d_in[0];
    const float* offset = (const float*)d_in[1];
    const float* w0     = (const float*)d_in[2];
    const float* b0     = (const float*)d_in[3];
    const float* w1     = (const float*)d_in[4];
    const float* b1     = (const float*)d_in[5];
    float* out          = (float*)d_out;

    char* ws = (char*)d_ws;
    const size_t QT_BYTES = (size_t)NPIX * 256 * 2;      // 33 554 432
    __hip_bfloat16* qt  = (__hip_bfloat16*)(ws);
    __hip_bfloat16* xt  = (__hip_bfloat16*)(ws + QT_BYTES);
    __hip_bfloat16* w0b = (__hip_bfloat16*)(ws + 2 * QT_BYTES);
    __hip_bfloat16* w1b = (__hip_bfloat16*)(ws + 2 * QT_BYTES + (size_t)C_ * C_ * 2);

    conv_w_kernel<<<(C_ * C_) / 256, 256, 0, stream>>>(w0, w1, w0b, w1b);
    transpose_x_kernel<<<B_ * (C_ / 64) * (HW_ / 64), 256, 0, stream>>>(x, xt);
    deform_max_kernel<<<(NPIX * 8) / 256, 256, 0, stream>>>(x, offset, qt);
    fused_gemm_kernel<<<NPIX / 16, 256, 0, stream>>>(qt, xt, w0b, w1b, b0, b1, out);
}
